// SIlayer_89558658056409
// MI455X (gfx1250) — matrix operand order repaired and checked by reading, not compiled, not run on hardware
//
#include <hip/hip_runtime.h>
#include <hip/hip_bf16.h>

typedef __bf16 bf16;
typedef __attribute__((ext_vector_type(16))) __bf16 v16bf;
typedef __attribute__((ext_vector_type(8)))  __bf16 v8bf;
typedef __attribute__((ext_vector_type(4)))  __bf16 v4bf;
typedef __attribute__((ext_vector_type(8)))  float   v8f;

#define WMMA_BF16(a, b, c) \
  __builtin_amdgcn_wmma_f32_16x16x32_bf16(false, (a), false, (b), (short)0, (c), false, false)

static __device__ __forceinline__ bf16 f2b(float f) { return (bf16)f; }

static __device__ __forceinline__ v8f zero8() {
  v8f z;
#pragma unroll
  for (int i = 0; i < 8; ++i) z[i] = 0.f;
  return z;
}

// ---- fragment loaders (wave32); A/B use contiguous k = 16*(lane/16) + s ----

// A-matrix 16x32 bf16 tile from row-major [rows][ld] at (row0, k0).
// Per lane: two contiguous 16-byte groups -> 2x b128 loads.
static __device__ __forceinline__ v16bf load_a_lds(const bf16* __restrict__ src, int ld,
                                                   int row0, int k0) {
  int l = threadIdx.x & 31;
  const bf16* p = src + (row0 + (l & 15)) * ld + k0 + ((l >> 4) & 1) * 16;
  v8bf g0 = *(const v8bf*)(p);
  v8bf g1 = *(const v8bf*)(p + 8);
  v16bf a;
#pragma unroll
  for (int i = 0; i < 8; ++i) { a[i] = g0[i]; a[8 + i] = g1[i]; }
  return a;
}

// B-matrix 32x16 tile from a K-contiguous (N-major, [N][ld]) source:
// B(k,n) = src[(n0+n)*ld + k0 + k]. Per lane: one contiguous 32-byte run.
static __device__ __forceinline__ v16bf load_b_kmaj(const bf16* __restrict__ src, int ld,
                                                    int k0, int n0) {
  int l = threadIdx.x & 31;
  const bf16* p = src + (n0 + (l & 15)) * ld + k0 + ((l >> 4) & 1) * 16;
  return *(const v16bf*)p;
}

// C/D 16x16 f32: lane = col (n), VGPR v = row (M = msel*8 + v).
static __device__ __forceinline__ void store_c_bf(bf16* dst, int ld, int row0, int n0, v8f c) {
  int l = threadIdx.x & 31;
  bf16* p = dst + (row0 + ((l >> 4) & 1) * 8) * ld + n0 + (l & 15);
#pragma unroll
  for (int v = 0; v < 8; ++v) p[v * ld] = f2b(c[v]);
}

// Transposed C store into an N-major [N][ld] buffer: per lane one packed 16-byte write.
static __device__ __forceinline__ void store_c_bf_tr(bf16* dst, int ld, int row0, int n0, v8f c) {
  int l = threadIdx.x & 31;
  bf16* p = dst + (n0 + (l & 15)) * ld + row0 + ((l >> 4) & 1) * 8;
  v8bf q;
#pragma unroll
  for (int v = 0; v < 8; ++v) q[v] = f2b(c[v]);
  *(v8bf*)p = q;
}

static __device__ __forceinline__ void store_c_f32(float* dst, int ld, int row0, int n0, v8f c) {
  int l = threadIdx.x & 31;
  float* p = dst + (row0 + ((l >> 4) & 1) * 8) * ld + n0 + (l & 15);
#pragma unroll
  for (int v = 0; v < 8; ++v) p[v * ld] = c[v];
}

// ---- cooperative LayerNorm over LDS [64][128] f32, 256 threads (4 per row) ----
static __device__ __forceinline__ void ln_inplace_bf(float* buf, const float* __restrict__ g,
                                                     const float* __restrict__ b, bf16* bfout) {
  int tid = threadIdx.x;
  int row = tid >> 2, part = tid & 3;
  float* r = buf + row * 128 + part * 32;
  float s = 0.f;
#pragma unroll
  for (int i = 0; i < 32; ++i) s += r[i];
  s += __shfl_xor(s, 1, 32);
  s += __shfl_xor(s, 2, 32);
  float mean = s * (1.0f / 128.0f);
  float vs = 0.f;
#pragma unroll
  for (int i = 0; i < 32; ++i) { float d = r[i] - mean; vs += d * d; }
  vs += __shfl_xor(vs, 1, 32);
  vs += __shfl_xor(vs, 2, 32);
  float rstd = rsqrtf(vs * (1.0f / 128.0f) + 1e-5f);
#pragma unroll
  for (int i = 0; i < 32; ++i) {
    int col = part * 32 + i;
    float y = (r[i] - mean) * rstd * g[col] + b[col];
    r[i] = y;
    bfout[row * 128 + col] = f2b(y);
  }
}

static __device__ __forceinline__ void ln_store_global(const float* buf, const float* __restrict__ g,
                                                       const float* __restrict__ b,
                                                       float* __restrict__ gout) {
  int tid = threadIdx.x;
  int row = tid >> 2, part = tid & 3;
  const float* r = buf + row * 128 + part * 32;
  float s = 0.f;
#pragma unroll
  for (int i = 0; i < 32; ++i) s += r[i];
  s += __shfl_xor(s, 1, 32);
  s += __shfl_xor(s, 2, 32);
  float mean = s * (1.0f / 128.0f);
  float vs = 0.f;
#pragma unroll
  for (int i = 0; i < 32; ++i) { float d = r[i] - mean; vs += d * d; }
  vs += __shfl_xor(vs, 1, 32);
  vs += __shfl_xor(vs, 2, 32);
  float rstd = rsqrtf(vs * (1.0f / 128.0f) + 1e-5f);
#pragma unroll
  for (int i = 0; i < 32; ++i) {
    int col = part * 32 + i;
    gout[row * 128 + col] = (r[i] - mean) * rstd * g[col] + b[col];
  }
}

// ---- weight bf16 arena offsets (elements); ALL matrices stored TRANSPOSED ([N][K]) ----
#define OFF_WQ  0         // [128][128]
#define OFF_WK  16384
#define OFF_WV  32768
#define OFF_WFC 49152
#define OFF_WSI 65536     // [128][512]
#define OFF_W1S 131072    // [512][128]
#define OFF_W2S 196608    // [128][512]
#define OFF_W1I 262144    // [512][128]
#define OFF_W2I 327680    // [128][512]

__global__ __launch_bounds__(256) void prep_kernel(
    const float* __restrict__ wq, const float* __restrict__ wk, const float* __restrict__ wv,
    const float* __restrict__ wfc, const float* __restrict__ wsi, const float* __restrict__ w1s,
    const float* __restrict__ w2s, const float* __restrict__ w1i, const float* __restrict__ w2i,
    float* __restrict__ att_sum, bf16* __restrict__ wb) {
  int i0 = blockIdx.x * blockDim.x + threadIdx.x;
  int stride = gridDim.x * blockDim.x;
  for (int k = i0; k < 128 * 4 * 64 * 64; k += stride) att_sum[k] = 0.f;
  // (128,128) mats -> transposed [n][k]
  for (int k = i0; k < 16384; k += stride) {
    int r = k >> 7, c = k & 127;
    wb[OFF_WQ  + c * 128 + r] = f2b(wq[k]);
    wb[OFF_WK  + c * 128 + r] = f2b(wk[k]);
    wb[OFF_WV  + c * 128 + r] = f2b(wv[k]);
    wb[OFF_WFC + c * 128 + r] = f2b(wfc[k]);
  }
  for (int k = i0; k < 65536; k += stride) {
    {  // (512,128) sources -> [128][512]
      int r = k >> 7, c = k & 127;
      wb[OFF_WSI + c * 512 + r] = f2b(wsi[k]);
      wb[OFF_W2S + c * 512 + r] = f2b(w2s[k]);
      wb[OFF_W2I + c * 512 + r] = f2b(w2i[k]);
    }
    {  // (128,512) sources -> [512][128]
      int r = k >> 9, c = k & 511;
      wb[OFF_W1S + c * 128 + r] = f2b(w1s[k]);
      wb[OFF_W1I + c * 128 + r] = f2b(w1i[k]);
    }
  }
}

// =============== Kernel 1: neighbor MHA + inter FFN, one block per (n,nn) ===============
__global__ __launch_bounds__(256) void neigh_attn_kernel(
    const float* __restrict__ nb_emb,         // (N,NN,S,D)
    const unsigned char* __restrict__ mask,   // (N,NN,S,S) bool bytes
    const bf16* __restrict__ wb,
    float* __restrict__ att_sum,              // (N,H,S,S), pre-zeroed
    const float* __restrict__ g_mha, const float* __restrict__ b_mha,
    const float* __restrict__ g_int, const float* __restrict__ b_int,
    float* __restrict__ out_nb)               // (N,NN,S,D)
{
  __shared__ __align__(32) unsigned char smem[65536];
  bf16*  r0b  = (bf16*)smem;             // 0..32K: Xbf (16K) then P[4][64][64] bf16
  float* r0f  = (float*)smem;            // 0..32K: f32 [64][128] (mha out / res_inter)
  bf16*  Qb   = (bf16*)(smem + 16384);   // 16..32K  [64][128] (A)
  bf16*  Kb   = (bf16*)(smem + 32768);   // 32..48K  [64][128] (K-major B source for K^T)
  bf16*  Vt   = (bf16*)(smem + 49152);   // 48..64K  [128][64] V transposed (B source)
  bf16*  resb = (bf16*)(smem + 32768);   // reuse Kb region (A)
  bf16*  hcb  = (bf16*)(smem + 49152);   // reuse Vt region (A)

  const int bid = blockIdx.x;
  const int n = bid >> 4, nn = bid & 15;
  const int tid = threadIdx.x;
  const int w = __builtin_amdgcn_readfirstlane(tid >> 5);  // wave id, provably uniform
  const int l = tid & 31;
  const int nlo = l & 15, msel = (l >> 4) & 1;

  const bf16* WQ  = wb + OFF_WQ;
  const bf16* WK  = wb + OFF_WK;
  const bf16* WV  = wb + OFF_WV;
  const bf16* WFC = wb + OFF_WFC;
  const bf16* W1I = wb + OFF_W1I;
  const bf16* W2I = wb + OFF_W2I;

  // warm caches for the weight streams (global_prefetch_b8)
  __builtin_prefetch(WQ, 0, 3);
  __builtin_prefetch(WK, 0, 3);
  __builtin_prefetch(WV, 0, 3);
  __builtin_prefetch(WFC, 0, 3);
  __builtin_prefetch(W1I, 0, 3);
  __builtin_prefetch(W2I, 0, 3);

  const float* Xg = nb_emb + ((size_t)(n * 16 + nn)) * 64 * 128;

  // phase 0: X -> bf16 LDS (float4 reads, packed v4bf writes)
  for (int i = tid; i < 2048; i += 256) {
    float4 fv = ((const float4*)Xg)[i];
    v4bf o;
    o[0] = f2b(fv.x); o[1] = f2b(fv.y); o[2] = f2b(fv.z); o[3] = f2b(fv.w);
    ((v4bf*)r0b)[i] = o;
  }
  __syncthreads();

  // phase 1: Q/K/V = X @ W{q,k,v}; one shared A fragment per k-step, 4 tiles/wave/matrix
#pragma unroll 1
  for (int j = 0; j < 4; ++j) {
    int t = w * 4 + j;
    int rb = t >> 3, ct = t & 7;
    v8f cq = zero8(), ck = zero8(), cv = zero8();
#pragma unroll
    for (int ks = 0; ks < 4; ++ks) {
      v16bf a = load_a_lds(r0b, 128, rb * 16, ks * 32);
      cq = WMMA_BF16(a, load_b_kmaj(WQ, 128, ks * 32, ct * 16), cq);
      ck = WMMA_BF16(a, load_b_kmaj(WK, 128, ks * 32, ct * 16), ck);
      cv = WMMA_BF16(a, load_b_kmaj(WV, 128, ks * 32, ct * 16), cv);
    }
    store_c_bf(Qb, 128, rb * 16, ct * 16, cq);
    store_c_bf(Kb, 128, rb * 16, ct * 16, ck);
    store_c_bf_tr(Vt, 64, rb * 16, ct * 16, cv);
  }
  __syncthreads();

  // phase 2: scores = QK^T/sqrt(DK), mask, softmax -> P bf16 (overwrites 0..32K)
  {
    v16bf aQ[2];
    v16bf bK[2][4];
#pragma unroll
    for (int aidx = 0; aidx < 2; ++aidx) {
      int idx = w * 2 + aidx;
      int h = idx >> 2, rb = idx & 3;
      aQ[aidx] = load_a_lds(Qb, 128, rb * 16, h * 32);
#pragma unroll
      for (int t = 0; t < 4; ++t) bK[aidx][t] = load_b_kmaj(Kb, 128, h * 32, t * 16);
    }
    __syncthreads();  // all Q/K fragment reads done -> region0/Qb free for P

    const unsigned char* mrow = mask + ((size_t)(n * 16 + nn)) * 64 * 64;
    const float scale = 0.17677669529663687f;  // 1/sqrt(32)
#pragma unroll
    for (int aidx = 0; aidx < 2; ++aidx) {
      int idx = w * 2 + aidx;
      int h = idx >> 2, rb = idx & 3;
      v8f s[4];
#pragma unroll
      for (int t = 0; t < 4; ++t) { v8f z = zero8(); s[t] = WMMA_BF16(aQ[aidx], bK[aidx][t], z); }
#pragma unroll
      for (int v = 0; v < 8; ++v) {
        int r = rb * 16 + msel * 8 + v;
        float sv[4];
        float mx = -3.4e38f;
#pragma unroll
        for (int t = 0; t < 4; ++t) {
          float x = s[t][v] * scale;
          if (!mrow[r * 64 + t * 16 + nlo]) x = -1e9f;
          sv[t] = x;
          mx = fmaxf(mx, x);
        }
        mx = fmaxf(mx, __shfl_xor(mx, 1, 32));
        mx = fmaxf(mx, __shfl_xor(mx, 2, 32));
        mx = fmaxf(mx, __shfl_xor(mx, 4, 32));
        mx = fmaxf(mx, __shfl_xor(mx, 8, 32));
        float sum = 0.f;
#pragma unroll
        for (int t = 0; t < 4; ++t) { sv[t] = __expf(sv[t] - mx); sum += sv[t]; }
        sum += __shfl_xor(sum, 1, 32);
        sum += __shfl_xor(sum, 2, 32);
        sum += __shfl_xor(sum, 4, 32);
        sum += __shfl_xor(sum, 8, 32);
        float inv = 1.0f / sum;
#pragma unroll
        for (int t = 0; t < 4; ++t) {
          float p = sv[t] * inv;
          int col = t * 16 + nlo;
          r0b[h * 4096 + r * 64 + col] = f2b(p);
          atomicAdd(&att_sum[(((size_t)n * 4 + h) * 64 + r) * 64 + col], p * 0.0625f);
        }
      }
    }
  }
  __syncthreads();

  // phase 3: res = P @ V -> resb (32..48K), B from transposed Vt
#pragma unroll
  for (int j = 0; j < 4; ++j) {
    int t = w * 4 + j;
    int rb = t >> 3, ct = t & 7;
    int h = ct >> 1;
    int n0 = h * 32 + (ct & 1) * 16;
    v8f c = zero8();
#pragma unroll
    for (int ks = 0; ks < 2; ++ks) {
      v16bf a = load_a_lds(r0b + h * 4096, 64, rb * 16, ks * 32);
      v16bf b = load_b_kmaj(Vt, 64, ks * 32, n0);
      c = WMMA_BF16(a, b, c);
    }
    store_c_bf(resb, 128, rb * 16, ct * 16, c);
  }
  __syncthreads();

  const int rbw = w >> 1;
  const int cbase = (w & 1) * 4;

  // phase 4: out = res @ W_fc + X (residual from global f32) -> region0 f32
#pragma unroll
  for (int j = 0; j < 4; ++j) {
    int ct = cbase + j;
    v8f c = zero8();
#pragma unroll
    for (int ks = 0; ks < 4; ++ks) {
      v16bf a = load_a_lds(resb, 128, rbw * 16, ks * 32);
      v16bf b = load_b_kmaj(WFC, 128, ks * 32, ct * 16);
      c = WMMA_BF16(a, b, c);
    }
#pragma unroll
    for (int v = 0; v < 8; ++v) {
      int r = rbw * 16 + msel * 8 + v;
      c[v] += Xg[r * 128 + ct * 16 + nlo];
    }
    store_c_f32(r0f, 128, rbw * 16, ct * 16, c);
  }
  __syncthreads();

  // LN(mha): in-place f32 + bf16 copy to resb (A-matrix for FFN)
  ln_inplace_bf(r0f, g_mha, b_mha, resb);
  __syncthreads();

  // phase 5: inter FFN, chunked over DFF (4 x 128), acc in C-frags
  {
    v8f acc[4];
#pragma unroll
    for (int j = 0; j < 4; ++j) acc[j] = zero8();
#pragma unroll 1
    for (int cc = 0; cc < 4; ++cc) {
#pragma unroll
      for (int j = 0; j < 4; ++j) {
        int ct = cbase + j;
        v8f c = zero8();
#pragma unroll
        for (int ks = 0; ks < 4; ++ks) {
          v16bf a = load_a_lds(resb, 128, rbw * 16, ks * 32);
          v16bf b = load_b_kmaj(W1I, 128, ks * 32, cc * 128 + ct * 16);
          c = WMMA_BF16(a, b, c);
        }
#pragma unroll
        for (int v = 0; v < 8; ++v) c[v] = fmaxf(c[v], 0.f);
        store_c_bf(hcb, 128, rbw * 16, ct * 16, c);
      }
      __syncthreads();
#pragma unroll
      for (int j = 0; j < 4; ++j) {
        int ct = cbase + j;
#pragma unroll
        for (int ks = 0; ks < 4; ++ks) {
          v16bf a = load_a_lds(hcb, 128, rbw * 16, ks * 32);
          v16bf b = load_b_kmaj(W2I, 512, cc * 128 + ks * 32, ct * 16);
          acc[j] = WMMA_BF16(a, b, acc[j]);
        }
      }
      __syncthreads();
    }
    // residual (per-lane same addresses as the store -> race-free) and stage for LN
#pragma unroll
    for (int j = 0; j < 4; ++j) {
      int ct = cbase + j;
#pragma unroll
      for (int v = 0; v < 8; ++v) {
        int r = rbw * 16 + msel * 8 + v;
        acc[j][v] += r0f[r * 128 + ct * 16 + nlo];
      }
      store_c_f32(r0f, 128, rbw * 16, ct * 16, acc[j]);
    }
  }
  __syncthreads();

  ln_store_global(r0f, g_int, b_int, out_nb + ((size_t)(n * 16 + nn)) * 64 * 128);
}

// =============== Kernel 2: self path, one block per agent n ===============
__global__ __launch_bounds__(256) void self_kernel(
    const float* __restrict__ x_emb,    // (N,S,D)
    const float* __restrict__ att_sum,  // (N,H,S,S), already includes 1/NN
    const float* __restrict__ att_factor,
    const bf16* __restrict__ wb,
    const float* __restrict__ g_self, const float* __restrict__ b_self,
    float* __restrict__ out_x)          // (N,S,D)
{
  __shared__ __align__(32) unsigned char smem[65536];
  bf16*  Xb   = (bf16*)smem;             // 0..16K [64][128] (A; also K-major B for X X^T)
  bf16*  Xt   = (bf16*)(smem + 16384);   // 16..32K [128][64] X transposed (B for att@X)
  bf16*  attS = (bf16*)(smem + 32768);   // 32..40K att_self bf16 [64][64]
  bf16*  Pb   = (bf16*)(smem + 40960);   // 40..48K P_h bf16 [64][64]
  bf16*  Rb   = (bf16*)(smem + 49152);   // 48..64K res_si bf16 [64][128]
  float* x1f  = (float*)(smem + 32768);  // 32..64K x1 f32 [64][128] (after head loop)
  bf16*  x1b  = (bf16*)(smem + 16384);   // 16..32K x1 bf16 (over Xt, after head loop)
  bf16*  hcb  = (bf16*)smem;             // 0..16K FFN hidden (over Xb)

  const int n = blockIdx.x;
  const int tid = threadIdx.x;
  const int w = __builtin_amdgcn_readfirstlane(tid >> 5);  // wave id, provably uniform
  const int l = tid & 31;
  const int nlo = l & 15, msel = (l >> 4) & 1;
  const float factor = att_factor[0];

  const bf16* WSI = wb + OFF_WSI;   // [128][512]
  const bf16* W1S = wb + OFF_W1S;   // [512][128]
  const bf16* W2S = wb + OFF_W2S;   // [128][512]

  __builtin_prefetch(WSI, 0, 3);
  __builtin_prefetch(W1S, 0, 3);
  __builtin_prefetch(W2S, 0, 3);

  const float* Xg = x_emb + (size_t)n * 64 * 128;
  for (int i = tid; i < 2048; i += 256) {
    float4 fv = ((const float4*)Xg)[i];
    v4bf o;
    o[0] = f2b(fv.x); o[1] = f2b(fv.y); o[2] = f2b(fv.z); o[3] = f2b(fv.w);
    ((v4bf*)Xb)[i] = o;
    int e = i * 4;
    int r = e >> 7, c = e & 127;
    Xt[(c + 0) * 64 + r] = o[0];
    Xt[(c + 1) * 64 + r] = o[1];
    Xt[(c + 2) * 64 + r] = o[2];
    Xt[(c + 3) * 64 + r] = o[3];
  }
  __syncthreads();

  // att_self = softmax(X X^T / sqrt(128)) -- waves 0..3, one row-block each (scalar branch)
  if (w < 4) {
    int rb = w;
    v8f s[4];
#pragma unroll
    for (int t = 0; t < 4; ++t) {
      v8f c = zero8();
#pragma unroll
      for (int ks = 0; ks < 4; ++ks) {
        v16bf a = load_a_lds(Xb, 128, rb * 16, ks * 32);
        v16bf b = load_b_kmaj(Xb, 128, ks * 32, t * 16);  // B(k,t)=X[t][k]
        c = WMMA_BF16(a, b, c);
      }
      s[t] = c;
    }
    const float scale = 0.08838834764831845f;  // 1/sqrt(128)
#pragma unroll
    for (int v = 0; v < 8; ++v) {
      float sv[4];
      float mx = -3.4e38f;
#pragma unroll
      for (int t = 0; t < 4; ++t) { sv[t] = s[t][v] * scale; mx = fmaxf(mx, sv[t]); }
      mx = fmaxf(mx, __shfl_xor(mx, 1, 32));
      mx = fmaxf(mx, __shfl_xor(mx, 2, 32));
      mx = fmaxf(mx, __shfl_xor(mx, 4, 32));
      mx = fmaxf(mx, __shfl_xor(mx, 8, 32));
      float sum = 0.f;
#pragma unroll
      for (int t = 0; t < 4; ++t) { sv[t] = __expf(sv[t] - mx); sum += sv[t]; }
      sum += __shfl_xor(sum, 1, 32);
      sum += __shfl_xor(sum, 2, 32);
      sum += __shfl_xor(sum, 4, 32);
      sum += __shfl_xor(sum, 8, 32);
      float inv = 1.0f / sum;
      int r = rb * 16 + msel * 8 + v;
#pragma unroll
      for (int t = 0; t < 4; ++t) attS[r * 64 + t * 16 + nlo] = f2b(sv[t] * inv);
    }
  }
  __syncthreads();

  const int rbw = w >> 1;
  const int cbase = (w & 1) * 4;
  const float* asum_n = att_sum + (size_t)n * 4 * 4096;

  // head loop: att = att_self + factor*att_i ; res_si = att @ X ; x1 += res_si @ W_SI[h]
  v8f xacc[4];
#pragma unroll
  for (int j = 0; j < 4; ++j) xacc[j] = zero8();
#pragma unroll 1
  for (int h = 0; h < 4; ++h) {
    for (int i = tid; i < 4096; i += 256)
      Pb[i] = f2b((float)attS[i] + factor * asum_n[h * 4096 + i]);
    __syncthreads();
#pragma unroll
    for (int j = 0; j < 4; ++j) {
      int ct = cbase + j;
      v8f c = zero8();
#pragma unroll
      for (int ks = 0; ks < 2; ++ks) {
        v16bf a = load_a_lds(Pb, 64, rbw * 16, ks * 32);
        v16bf b = load_b_kmaj(Xt, 64, ks * 32, ct * 16);  // B(k,d)=X[k][d]=Xt[d][k]
        c = WMMA_BF16(a, b, c);
      }
      store_c_bf(Rb, 128, rbw * 16, ct * 16, c);
    }
    __syncthreads();
#pragma unroll
    for (int j = 0; j < 4; ++j) {
      int ct = cbase + j;
#pragma unroll
      for (int ks = 0; ks < 4; ++ks) {
        v16bf a = load_a_lds(Rb, 128, rbw * 16, ks * 32);
        v16bf b = load_b_kmaj(WSI, 512, h * 128 + ks * 32, ct * 16);
        xacc[j] = WMMA_BF16(a, b, xacc[j]);
      }
    }
    __syncthreads();
  }
#pragma unroll
  for (int j = 0; j < 4; ++j) {
    int ct = cbase + j;
    store_c_f32(x1f, 128, rbw * 16, ct * 16, xacc[j]);
    store_c_bf(x1b, 128, rbw * 16, ct * 16, xacc[j]);
  }
  __syncthreads();

  // self FFN, chunked
  {
    v8f acc[4];
#pragma unroll
    for (int j = 0; j < 4; ++j) acc[j] = zero8();
#pragma unroll 1
    for (int cc = 0; cc < 4; ++cc) {
#pragma unroll
      for (int j = 0; j < 4; ++j) {
        int ct = cbase + j;
        v8f c = zero8();
#pragma unroll
        for (int ks = 0; ks < 4; ++ks) {
          v16bf a = load_a_lds(x1b, 128, rbw * 16, ks * 32);
          v16bf b = load_b_kmaj(W1S, 128, ks * 32, cc * 128 + ct * 16);
          c = WMMA_BF16(a, b, c);
        }
#pragma unroll
        for (int v = 0; v < 8; ++v) c[v] = fmaxf(c[v], 0.f);
        store_c_bf(hcb, 128, rbw * 16, ct * 16, c);
      }
      __syncthreads();
#pragma unroll
      for (int j = 0; j < 4; ++j) {
        int ct = cbase + j;
#pragma unroll
        for (int ks = 0; ks < 4; ++ks) {
          v16bf a = load_a_lds(hcb, 128, rbw * 16, ks * 32);
          v16bf b = load_b_kmaj(W2S, 512, cc * 128 + ks * 32, ct * 16);
          acc[j] = WMMA_BF16(a, b, acc[j]);
        }
      }
      __syncthreads();
    }
#pragma unroll
    for (int j = 0; j < 4; ++j) {
      int ct = cbase + j;
#pragma unroll
      for (int v = 0; v < 8; ++v) {
        int r = rbw * 16 + msel * 8 + v;
        acc[j][v] += x1f[r * 128 + ct * 16 + nlo];
      }
      store_c_f32(x1f, 128, rbw * 16, ct * 16, acc[j]);
    }
  }
  __syncthreads();

  ln_store_global(x1f, g_self, b_self, out_x + (size_t)n * 64 * 128);
}

extern "C" void kernel_launch(void* const* d_in, const int* in_sizes, int n_in,
                              void* d_out, int out_size, void* d_ws, size_t ws_size,
                              hipStream_t stream) {
  (void)in_sizes; (void)n_in; (void)out_size; (void)ws_size;
  const float* x_emb  = (const float*)d_in[0];
  const float* nb_emb = (const float*)d_in[1];
  const unsigned char* mask = (const unsigned char*)d_in[2];
  const float* att_factor = (const float*)d_in[3];
  const float* W_Q  = (const float*)d_in[4];
  const float* W_K  = (const float*)d_in[5];
  const float* W_V  = (const float*)d_in[6];
  const float* W_fc = (const float*)d_in[7];
  const float* g_mha = (const float*)d_in[8];
  const float* b_mha = (const float*)d_in[9];
  const float* W_SI  = (const float*)d_in[10];
  const float* W1s   = (const float*)d_in[11];
  const float* W2s   = (const float*)d_in[12];
  const float* g_self = (const float*)d_in[13];
  const float* b_self = (const float*)d_in[14];
  const float* W1i   = (const float*)d_in[15];
  const float* W2i   = (const float*)d_in[16];
  const float* g_int = (const float*)d_in[17];
  const float* b_int = (const float*)d_in[18];

  float* att_sum = (float*)d_ws;                                  // 128*4*64*64 f32 = 8 MB
  bf16*  wb = (bf16*)((char*)d_ws + (size_t)(128 * 4 * 64 * 64) * sizeof(float));

  float* out_x  = (float*)d_out;                                  // (N,S,D)
  float* out_nb = out_x + 128 * 64 * 128;                         // (N,NN,S,D)

  prep_kernel<<<1024, 256, 0, stream>>>(W_Q, W_K, W_V, W_fc, W_SI, W1s, W2s, W1i, W2i,
                                        att_sum, wb);
  neigh_attn_kernel<<<2048, 256, 0, stream>>>(nb_emb, mask, wb, att_sum,
                                              g_mha, b_mha, g_int, b_int, out_nb);
  self_kernel<<<128, 256, 0, stream>>>(x_emb, att_sum, att_factor, wb,
                                       g_self, b_self, out_x);
}
